// EdgeModel_5188320494484
// MI455X (gfx1250) — matrix operand order repaired and checked by reading, not compiled, not run on hardware
//
#include <hip/hip_runtime.h>

typedef __attribute__((ext_vector_type(16))) _Float16 v16h;
typedef __attribute__((ext_vector_type(8)))  _Float16 v8h;
typedef __attribute__((ext_vector_type(8)))  float    v8f;

#if __has_builtin(__builtin_amdgcn_global_load_async_to_lds_b128) && \
    __has_builtin(__builtin_amdgcn_s_wait_asynccnt)
#define USE_ASYNC_LDS 1
#endif

// Shapes: E=655360, F=128 each segment, d_in=512, H=256, F_out=128.
// Layer1: [E,512]@[512,256]; Layer2: [E,256]@[256,128].

// ---------------------------------------------------------------------------
// Weight prep: fp32 -> f16, pre-swizzled into WMMA B-fragment lane layout,
// *kk-major* so each K-step tile is one contiguous chunk for LDS staging.
//   W1p[((kk*16 + nt)*32 + lane)*16 + i] = W1[K,N],
//     K = 32*kk + 8*(lane>>4) + (i&7) + 16*(i>>3),  N = 16*nt + (lane&15)
// ---------------------------------------------------------------------------
__global__ __launch_bounds__(256) void prep_w1(const float* __restrict__ W1,
                                               _Float16* __restrict__ W1p) {
  int tid  = blockIdx.x * 256 + threadIdx.x;   // 131072 total
  int i    = tid & 15;
  int lane = (tid >> 4) & 31;
  int nt   = (tid >> 9) & 15;
  int kk   = tid >> 13;
  int K    = 32 * kk + 8 * (lane >> 4) + (i & 7) + 16 * (i >> 3);   // 0..511
  int N    = 16 * nt + (lane & 15);            // 0..255
  W1p[tid] = (_Float16)W1[K * 256 + N];
}

__global__ __launch_bounds__(256) void prep_w2(const float* __restrict__ W2,
                                               _Float16* __restrict__ W2p) {
  int tid  = blockIdx.x * 256 + threadIdx.x;   // 32768 total
  int i    = tid & 15;
  int lane = (tid >> 4) & 31;
  int nt   = (tid >> 9) & 7;
  int kk   = tid >> 12;
  int K    = 32 * kk + 8 * (lane >> 4) + (i & 7) + 16 * (i >> 3);   // 0..255
  int N    = 16 * nt + (lane & 15);            // 0..127
  W2p[tid] = (_Float16)W2[K * 128 + N];
}

// ---------------------------------------------------------------------------
// Fused MLP: 128 threads = 4 waves; 64 edges per block, 16 per wave.
// B tiles double-buffered in LDS, shared by all 4 waves.
// ---------------------------------------------------------------------------
__global__ __launch_bounds__(128) void edge_mlp(
    const float* __restrict__ src, const float* __restrict__ dst,
    const float* __restrict__ ea,  const float* __restrict__ u,
    const int*   __restrict__ batch,
    const float* __restrict__ b1,  const float* __restrict__ b2,
    const _Float16* __restrict__ W1p, const _Float16* __restrict__ W2p,
    float* __restrict__ out)
{
  __shared__ __align__(64) _Float16 bbuf[2][8192];      // 2 x 16 KB B tiles
  __shared__ __align__(64) _Float16 hbuf[4][16][256];   // 32 KB hidden acts

  const int t    = threadIdx.x;
  const int lane = t & 31;
  const int w    = t >> 5;
  const int hh   = lane >> 4;
  const int ln   = lane & 15;
  const int m0   = blockIdx.x * 64 + w * 16;
  const int row  = m0 + ln;

  const float* segp[4] = { src + row * 128,
                           dst + row * 128,
                           ea  + row * 128,
                           u + batch[row] * 128 };

  // ---- staging helpers: copy `chunks` x 2KB from g into l, block-wide ----
#ifdef USE_ASYNC_LDS
  typedef int v4i __attribute__((vector_size(16)));
  typedef __attribute__((address_space(1))) v4i* gvp;   // global int4*
  typedef __attribute__((address_space(3))) v4i* lvp;   // LDS int4*
  auto stage_issue = [&](const _Float16* g, _Float16* l, int chunks, float4*) {
    for (int r = 0; r < chunks; ++r)
      __builtin_amdgcn_global_load_async_to_lds_b128(
          (gvp)(v4i*)(g + r * 1024 + t * 8),
          (lvp)(v4i*)(l + r * 1024 + t * 8), 0, 0);
  };
  auto stage_commit = [&](float4*, _Float16*, int) {
    __builtin_amdgcn_s_wait_asynccnt(0);
  };
#else
  auto stage_issue = [&](const _Float16* g, _Float16*, int chunks, float4* rr) {
    for (int r = 0; r < chunks; ++r)
      rr[r] = *((const float4*)g + r * 128 + t);
  };
  auto stage_commit = [&](float4* rr, _Float16* l, int chunks) {
    for (int r = 0; r < chunks; ++r)
      *((float4*)l + r * 128 + t) = rr[r];
  };
#endif

  auto a_issue = [&](int kk, float4* r) {
    const float* p = segp[kk >> 2] + (kk & 3) * 32 + 8 * hh;
    r[0] = *(const float4*)(p + 0);
    r[1] = *(const float4*)(p + 4);
    r[2] = *(const float4*)(p + 16);
    r[3] = *(const float4*)(p + 20);
  };
  auto a_pack = [&](const float4* r) {
    v16h a;
    a[0]=(_Float16)r[0].x;  a[1]=(_Float16)r[0].y;  a[2]=(_Float16)r[0].z;  a[3]=(_Float16)r[0].w;
    a[4]=(_Float16)r[1].x;  a[5]=(_Float16)r[1].y;  a[6]=(_Float16)r[1].z;  a[7]=(_Float16)r[1].w;
    a[8]=(_Float16)r[2].x;  a[9]=(_Float16)r[2].y;  a[10]=(_Float16)r[2].z; a[11]=(_Float16)r[2].w;
    a[12]=(_Float16)r[3].x; a[13]=(_Float16)r[3].y; a[14]=(_Float16)r[3].z; a[15]=(_Float16)r[3].w;
    return a;
  };

  const v8f zero = {0.f,0.f,0.f,0.f,0.f,0.f,0.f,0.f};
  float4 areg[4], anext[4], breg[8];

  // ================= Layer 1: [16x512] @ [512x256] =================
  a_issue(0, areg);
  stage_issue(W1p, bbuf[0], 8, breg);
  stage_commit(breg, bbuf[0], 8);
  __syncthreads();

  v8f acc[16];
#pragma unroll
  for (int nt = 0; nt < 16; ++nt) acc[nt] = zero;

#pragma unroll
  for (int kk = 0; kk < 16; ++kk) {
    if (kk < 15) {
      a_issue(kk + 1, anext);
      stage_issue(W1p + (kk + 1) * 8192, bbuf[(kk + 1) & 1], 8, breg);
    }
    v16h a = a_pack(areg);
    const _Float16* bl = bbuf[kk & 1];
#pragma unroll
    for (int nt = 0; nt < 16; ++nt) {
      v16h b = *(const v16h*)(bl + (nt * 32 + lane) * 16);
      acc[nt] = __builtin_amdgcn_wmma_f32_16x16x32_f16(
          false, a, false, b, (short)0, acc[nt], false, false);
    }
    if (kk < 15) stage_commit(breg, bbuf[(kk + 1) & 1], 8);
    __syncthreads();
    if (kk < 15) {
#pragma unroll
      for (int q = 0; q < 4; ++q) areg[q] = anext[q];
    }
  }

  // bias + ReLU -> per-wave h slice (f16).  acc[nt][j] is (row j+8*hh, col 16*nt+ln)
#pragma unroll
  for (int nt = 0; nt < 16; ++nt) {
    float bias = b1[nt * 16 + ln];
#pragma unroll
    for (int j = 0; j < 8; ++j) {
      float v = acc[nt][j] + bias;
      hbuf[w][j + 8 * hh][nt * 16 + ln] = (_Float16)(v > 0.f ? v : 0.f);
    }
  }

  // ================= Layer 2: [16x256] @ [256x128] =================
  stage_issue(W2p, bbuf[0], 4, breg);
  stage_commit(breg, bbuf[0], 4);
  __syncthreads();

  v8f acc2[8];
#pragma unroll
  for (int nt = 0; nt < 8; ++nt) acc2[nt] = zero;

#pragma unroll
  for (int kk = 0; kk < 8; ++kk) {
    if (kk < 7)
      stage_issue(W2p + (kk + 1) * 4096, bbuf[(kk + 1) & 1], 4, breg);

    v8h lo = *(const v8h*)&hbuf[w][ln][32 * kk + 8 * hh];
    v8h hi = *(const v8h*)&hbuf[w][ln][32 * kk + 16 + 8 * hh];
    v16h a;
#pragma unroll
    for (int i = 0; i < 8; ++i) { a[i] = lo[i]; a[8 + i] = hi[i]; }

    const _Float16* bl = bbuf[kk & 1];
#pragma unroll
    for (int nt = 0; nt < 8; ++nt) {
      v16h b = *(const v16h*)(bl + (nt * 32 + lane) * 16);
      acc2[nt] = __builtin_amdgcn_wmma_f32_16x16x32_f16(
          false, a, false, b, (short)0, acc2[nt], false, false);
    }
    if (kk < 7) stage_commit(breg, bbuf[(kk + 1) & 1], 4);
    __syncthreads();
  }

  // bias + fp32 store: acc2[nt][j] -> out[m0 + j + 8*hh][16*nt + ln]
#pragma unroll
  for (int nt = 0; nt < 8; ++nt) {
    float bias = b2[nt * 16 + ln];
#pragma unroll
    for (int j = 0; j < 8; ++j)
      out[(m0 + j + 8 * hh) * 128 + nt * 16 + ln] = acc2[nt][j] + bias;
  }
}

extern "C" void kernel_launch(void* const* d_in, const int* in_sizes, int n_in,
                              void* d_out, int out_size, void* d_ws, size_t ws_size,
                              hipStream_t stream) {
  const float* src   = (const float*)d_in[0];
  const float* dst   = (const float*)d_in[1];
  const float* ea    = (const float*)d_in[2];
  const float* u     = (const float*)d_in[3];
  const int*   batch = (const int*)d_in[4];
  const float* W1    = (const float*)d_in[5];
  const float* b1    = (const float*)d_in[6];
  const float* W2    = (const float*)d_in[7];
  const float* b2    = (const float*)d_in[8];
  float* out = (float*)d_out;

  _Float16* W1p = (_Float16*)d_ws;                              // 256 KB
  _Float16* W2p = (_Float16*)((char*)d_ws + 512 * 256 * 2);     //  64 KB

  prep_w1<<<131072 / 256, 256, 0, stream>>>(W1, W1p);
  prep_w2<<< 32768 / 256, 256, 0, stream>>>(W2, W2p);

  const int E = in_sizes[0] / 128;          // 655360
  edge_mlp<<<E / 64, 128, 0, stream>>>(src, dst, ea, u, batch, b1, b2,
                                       W1p, W2p, out);
}
